// mLSTMblock_74388833567240
// MI455X (gfx1250) — matrix operand order repaired and checked by reading, not compiled, not run on hardware
//
#include <hip/hip_runtime.h>
#include <hip/hip_bf16.h>
#include <cstdint>

// Problem dims (from reference)
#define B_     4
#define S_     16
#define D_     1024
#define H_     2048
#define ROWS_  64      // B*S
#define BLK_   256     // block-diagonal block size (H/DEPTH)

typedef __attribute__((ext_vector_type(16))) __bf16 v16bf;
typedef __attribute__((ext_vector_type(8)))  float  v8f;

// ---------------------------------------------------------------------------
// bf16-WMMA GEMM: C[m,n] = act(A[64,K] @ W[K,N] + bias[n]) (+resid)
// One workgroup (4 waves) per 16-wide N tile. Each wave owns one 16-row M
// tile (M=64 total). The 32x16 fp32 B tile is staged cooperatively into LDS
// with global_load_async_to_lds_b128 (each lane one 16B chunk), so weights
// cross the memory hierarchy once per N tile instead of once per (M,N) tile.
// blk > 0 => W is block-diagonal [DEPTH, blk, blk]; A columns offset by d*blk.
// act: 0=none, 1=silu, 2=exp, 3=sigmoid
// ---------------------------------------------------------------------------
__global__ __launch_bounds__(128) void gemm_bf16_wmma(
    const float* __restrict__ A, int lda,
    const float* __restrict__ W,
    const float* __restrict__ bias,
    const float* __restrict__ resid,
    float* __restrict__ C, int ldc,
    int N, int K, int act, int blk)
{
  __shared__ float tileB[32][16];   // [k][n] fp32 staging tile (2 KB)

  const int n0    = blockIdx.x << 4;
  const int wave  = (int)threadIdx.x >> 5;   // 0..3 -> M tile
  const int lane  = (int)threadIdx.x & 31;
  const int lhalf = lane >> 4;               // 0 or 1
  const int l     = lane & 15;
  const int m0    = wave << 4;

  const float* Wp = W;
  int acol0 = 0;
  int ldw   = N;
  int nw    = n0;
  if (blk > 0) {                   // block-diagonal weights
    const int d = n0 / blk;
    acol0 = d * blk;
    Wp    = W + (size_t)d * blk * blk;
    ldw   = blk;
    nw    = n0 - d * blk;
  }

  // Cooperative staging assignment: chunk = wave*32 + lane covers the whole
  // 32x16 tile in 16-byte pieces (row br, 4-float column chunk bc).
  const int ch = (wave << 5) + lane;         // 0..127
  const int br = ch >> 2;                    // B row (k offset) 0..31
  const int bc = (ch & 3) << 2;              // B col chunk: 0,4,8,12
  const unsigned ldsdst = (unsigned)(uintptr_t)&tileB[br][bc];

  const float* arow = A + (size_t)(m0 + l) * lda + acol0;

  v8f acc = {};
  for (int k0 = 0; k0 < K; k0 += 32) {
    // --- async-stage this K-step's B tile into LDS ---
    const float* src = Wp + (size_t)(k0 + br) * ldw + nw + bc;
    asm volatile("global_load_async_to_lds_b128 %0, %1, off"
                 :: "v"(ldsdst), "v"((unsigned long long)(uintptr_t)src)
                 : "memory");
    if (k0 + 32 < K)
      __builtin_prefetch((const void*)(src + (size_t)32 * ldw), 0, 0);

    // --- A fragment (ISA 7.12.2 bf16-A layout): per lane two contiguous
    //     8-float runs -> four 128-bit global loads ---
    const float4* a0 = (const float4*)(arow + k0 + (lhalf << 3));
    const float4* a1 = (const float4*)(arow + k0 + 16 + (lhalf << 3));
    const float4 r0 = a0[0], r1 = a0[1], r2 = a1[0], r3 = a1[1];
    v16bf a, b;
    a[0]  = (__bf16)r0.x; a[1]  = (__bf16)r0.y; a[2]  = (__bf16)r0.z; a[3]  = (__bf16)r0.w;
    a[4]  = (__bf16)r1.x; a[5]  = (__bf16)r1.y; a[6]  = (__bf16)r1.z; a[7]  = (__bf16)r1.w;
    a[8]  = (__bf16)r2.x; a[9]  = (__bf16)r2.y; a[10] = (__bf16)r2.z; a[11] = (__bf16)r2.w;
    a[12] = (__bf16)r3.x; a[13] = (__bf16)r3.y; a[14] = (__bf16)r3.z; a[15] = (__bf16)r3.w;

    // Wait for own async transfer, then barrier so all waves' chunks land.
    asm volatile("s_wait_asynccnt 0x0" ::: "memory");
    __syncthreads();

    // --- B fragment: lane = column l; lanes 0-15 take K=k0+0..7,16..23,
    //     lanes 16-31 take K=k0+8..15,24..31 ---
#pragma unroll
    for (int e = 0; e < 16; ++e)
      b[e] = (__bf16)tileB[(e < 8 ? (lhalf << 3) + e : 16 + (lhalf << 3) + (e - 8))][l];

    acc = __builtin_amdgcn_wmma_f32_16x16x32_bf16(
        false, a, false, b, (short)0, acc, false, false);

    __syncthreads();   // tile consumed by all waves before next overwrite
  }

  // C/D layout: lane = column (l), VGPR v -> row v + 8*lhalf
#pragma unroll
  for (int v = 0; v < 8; ++v) {
    const int row = m0 + v + (lhalf << 3);
    const int col = n0 + l;
    float r = acc[v] + bias[col];
    if (act == 1)      r = r * __builtin_amdgcn_rcpf(1.0f + __expf(-r));  // silu
    else if (act == 2) r = __expf(r);                                     // exp
    else if (act == 3) r = __builtin_amdgcn_rcpf(1.0f + __expf(-r));      // sigmoid
    if (resid) r += resid[(size_t)row * ldc + col];
    C[(size_t)row * ldc + col] = r;
  }
}

// ---------------------------------------------------------------------------
// Row layernorm: y[r, :] = (x - mu) * rsqrt(var + 1e-5) * g + b   (n <= 2048)
// ---------------------------------------------------------------------------
__global__ __launch_bounds__(256) void layernorm_rows(
    const float* __restrict__ x, const float* __restrict__ g,
    const float* __restrict__ b, float* __restrict__ y, int n)
{
  __shared__ float red[256];
  __shared__ float row[H_];
  const int r   = blockIdx.x;
  const int tid = threadIdx.x;
  const float* xr = x + (size_t)r * n;

  float s = 0.f;
  for (int j = tid; j < n; j += 256) { float v = xr[j]; row[j] = v; s += v; }
  red[tid] = s; __syncthreads();
  for (int off = 128; off > 0; off >>= 1) {
    if (tid < off) red[tid] += red[tid + off];
    __syncthreads();
  }
  const float mu = red[0] / n;
  __syncthreads();

  float s2 = 0.f;
  for (int j = tid; j < n; j += 256) { float d = row[j] - mu; s2 += d * d; }
  red[tid] = s2; __syncthreads();
  for (int off = 128; off > 0; off >>= 1) {
    if (tid < off) red[tid] += red[tid + off];
    __syncthreads();
  }
  const float rstd = rsqrtf(red[0] / n + 1e-5f);

  for (int j = tid; j < n; j += 256)
    y[(size_t)r * n + j] = (row[j] - mu) * rstd * g[j] + b[j];
}

// ---------------------------------------------------------------------------
// Causal conv over the hidden axis: one block per (b, o) pair.
// conv[b,o,t] = conv_b[o] + sum_i sum_{j<=t} left[b,i,j]*w[o,i, H-1-t+j]
// out = silu(conv)
// ---------------------------------------------------------------------------
__global__ __launch_bounds__(256) void causal_conv_silu(
    const float* __restrict__ left, const float* __restrict__ w,
    const float* __restrict__ cb, float* __restrict__ out)
{
  const int bo = blockIdx.x;   // b*S + o
  const int b  = bo >> 4;
  const int o  = bo & 15;
  for (int t = (int)threadIdx.x; t < H_; t += 256) {
    float acc = cb[o];
    for (int i = 0; i < S_; ++i) {
      const float* lrow = left + (size_t)(b * S_ + i) * H_;
      const float* wrow = w    + (size_t)(o * S_ + i) * H_ + (H_ - 1 - t);
      for (int j = 0; j <= t; ++j) acc += lrow[j] * wrow[j];
    }
    out[(size_t)bo * H_ + t] = acc * __builtin_amdgcn_rcpf(1.0f + __expf(-acc));
  }
}

// ---------------------------------------------------------------------------
// Elementwise mLSTM gate combine; emits p = ct*q and per-block max of nt*q.
// ---------------------------------------------------------------------------
__global__ __launch_bounds__(256) void gates_combine(
    const float* __restrict__ ig, const float* __restrict__ fg,
    const float* __restrict__ q,  const float* __restrict__ k,
    const float* __restrict__ v,  const float* __restrict__ ct1,
    const float* __restrict__ nt1,
    float* __restrict__ p, float* __restrict__ pmax)
{
  __shared__ float red[256];
  const int idx = blockIdx.x * 256 + (int)threadIdx.x;
  const float fi = fg[idx], ii = ig[idx], ki = k[idx], qi = q[idx];
  const float ct = fi * ct1[idx] + ii * v[idx] * ki;
  const float nt = fi * nt1[idx] + ii * ki;
  p[idx] = ct * qi;
  red[threadIdx.x] = nt * qi;
  __syncthreads();
  for (int off = 128; off > 0; off >>= 1) {
    if ((int)threadIdx.x < off)
      red[threadIdx.x] = fmaxf(red[threadIdx.x], red[threadIdx.x + off]);
    __syncthreads();
  }
  if (threadIdx.x == 0) pmax[blockIdx.x] = red[0];
}

__global__ __launch_bounds__(256) void max_reduce(
    const float* __restrict__ pmax, int n, float* __restrict__ g)
{
  __shared__ float red[256];
  float m = -3.402823466e38f;
  for (int i = (int)threadIdx.x; i < n; i += 256) m = fmaxf(m, pmax[i]);
  red[threadIdx.x] = m; __syncthreads();
  for (int off = 128; off > 0; off >>= 1) {
    if ((int)threadIdx.x < off)
      red[threadIdx.x] = fmaxf(red[threadIdx.x], red[threadIdx.x + off]);
    __syncthreads();
  }
  if (threadIdx.x == 0) g[0] = red[0];
}

// ---------------------------------------------------------------------------
// ht = o * p / gmax; left_out = LN(ht)*gn_g + gn_b + lskip; pre = left_out*right
// One block per row of H_.
// ---------------------------------------------------------------------------
__global__ __launch_bounds__(256) void out_norm_fused(
    const float* __restrict__ og, const float* __restrict__ p,
    const float* __restrict__ gmax,
    const float* __restrict__ gn_g, const float* __restrict__ gn_b,
    const float* __restrict__ lskip, const float* __restrict__ right,
    float* __restrict__ pre)
{
  __shared__ float red[256];
  __shared__ float row[H_];
  const int r   = blockIdx.x;
  const int tid = (int)threadIdx.x;
  const float inv = 1.0f / gmax[0];
  const size_t base = (size_t)r * H_;

  float s = 0.f;
  for (int j = tid; j < H_; j += 256) {
    float ht = og[base + j] * p[base + j] * inv;
    row[j] = ht; s += ht;
  }
  red[tid] = s; __syncthreads();
  for (int off = 128; off > 0; off >>= 1) {
    if (tid < off) red[tid] += red[tid + off];
    __syncthreads();
  }
  const float mu = red[0] / H_;
  __syncthreads();

  float s2 = 0.f;
  for (int j = tid; j < H_; j += 256) { float d = row[j] - mu; s2 += d * d; }
  red[tid] = s2; __syncthreads();
  for (int off = 128; off > 0; off >>= 1) {
    if (tid < off) red[tid] += red[tid + off];
    __syncthreads();
  }
  const float rstd = rsqrtf(red[0] / H_ + 1e-5f);

  for (int j = tid; j < H_; j += 256) {
    float lo = (row[j] - mu) * rstd * gn_g[j] + gn_b[j] + lskip[base + j];
    pre[base + j] = lo * right[base + j];
  }
}

// ---------------------------------------------------------------------------
extern "C" void kernel_launch(void* const* d_in, const int* in_sizes, int n_in,
                              void* d_out, int out_size, void* d_ws, size_t ws_size,
                              hipStream_t stream)
{
  const float* x       = (const float*)d_in[0];
  const float* ln_g    = (const float*)d_in[1];
  const float* ln_b    = (const float*)d_in[2];
  const float* left_w  = (const float*)d_in[3];
  const float* left_b  = (const float*)d_in[4];
  const float* right_w = (const float*)d_in[5];
  const float* right_b = (const float*)d_in[6];
  const float* conv_w  = (const float*)d_in[7];
  const float* conv_b  = (const float*)d_in[8];
  const float* lskip_w = (const float*)d_in[9];
  const float* lskip_b = (const float*)d_in[10];
  const float* wq_w    = (const float*)d_in[11];
  const float* wq_b    = (const float*)d_in[12];
  const float* wk_w    = (const float*)d_in[13];
  const float* wk_b    = (const float*)d_in[14];
  const float* wv_w    = (const float*)d_in[15];
  const float* wv_b    = (const float*)d_in[16];
  const float* ig_w    = (const float*)d_in[17];
  const float* ig_b    = (const float*)d_in[18];
  const float* fg_w    = (const float*)d_in[19];
  const float* fg_b    = (const float*)d_in[20];
  const float* og_w    = (const float*)d_in[21];
  const float* og_b    = (const float*)d_in[22];
  const float* gn_g    = (const float*)d_in[23];
  const float* gn_b    = (const float*)d_in[24];
  const float* proj_w  = (const float*)d_in[25];
  const float* proj_b  = (const float*)d_in[26];
  const float* ct_1    = (const float*)d_in[27];
  const float* nt_1    = (const float*)d_in[28];

  float* ws = (float*)d_ws;
  const size_t RD = (size_t)ROWS_ * D_;   // 65536
  const size_t RH = (size_t)ROWS_ * H_;   // 131072
  float* xn   = ws;            // [64,1024]
  float* left = xn   + RD;     // [64,2048]
  float* right= left + RH;
  float* ll   = right+ RH;     // silu(conv) = left_left
  float* lsk  = ll   + RH;
  float* q    = lsk  + RH;
  float* k    = q    + RH;
  float* v    = k    + RH;
  float* ig   = v    + RH;
  float* fg   = ig   + RH;
  float* og   = fg   + RH;
  float* p    = og   + RH;     // ct*q
  float* pre  = p    + RH;     // (LN(ht)+lskip)*right
  float* pmax = pre  + RH;     // 512 per-block maxes
  float* gmax = pmax + 512;    // 1

  const dim3 gH(H_ / 16);   // H-wide GEMMs: 128 N tiles x 4 waves
  const dim3 gD(D_ / 16);   // D-wide GEMMs

  // 1. x_n = LN(x)
  layernorm_rows<<<ROWS_, 256, 0, stream>>>(x, ln_g, ln_b, xn, D_);
  // 2. left / right = x_n @ W (+silu on right)
  gemm_bf16_wmma<<<gH, 128, 0, stream>>>(xn, D_, left_w,  left_b,  nullptr, left,  H_, H_, D_, 0, 0);
  gemm_bf16_wmma<<<gH, 128, 0, stream>>>(xn, D_, right_w, right_b, nullptr, right, H_, H_, D_, 1, 0);
  // 3. left_left = silu(causal_conv(left))
  causal_conv_silu<<<ROWS_, 256, 0, stream>>>(left, conv_w, conv_b, ll);
  // 4. l_skip and gate GEMMs
  gemm_bf16_wmma<<<gH, 128, 0, stream>>>(ll, H_, lskip_w, lskip_b, nullptr, lsk, H_, H_, H_, 0, 0);
  gemm_bf16_wmma<<<gH, 128, 0, stream>>>(ll,   H_, wq_w, wq_b, nullptr, q, H_, H_, BLK_, 0, BLK_);
  gemm_bf16_wmma<<<gH, 128, 0, stream>>>(ll,   H_, wk_w, wk_b, nullptr, k, H_, H_, BLK_, 0, BLK_);
  gemm_bf16_wmma<<<gH, 128, 0, stream>>>(left, H_, wv_w, wv_b, nullptr, v, H_, H_, BLK_, 0, BLK_);
  gemm_bf16_wmma<<<gH, 128, 0, stream>>>(ll, H_, ig_w, ig_b, nullptr, ig, H_, H_, H_, 2, 0);
  gemm_bf16_wmma<<<gH, 128, 0, stream>>>(ll, H_, fg_w, fg_b, nullptr, fg, H_, H_, H_, 2, 0);
  gemm_bf16_wmma<<<gH, 128, 0, stream>>>(ll, H_, og_w, og_b, nullptr, og, H_, H_, H_, 3, 0);
  // 5. gate combine + global max of nt*q
  gates_combine<<<(int)(RH / 256), 256, 0, stream>>>(ig, fg, q, k, v, ct_1, nt_1, p, pmax);
  max_reduce<<<1, 256, 0, stream>>>(pmax, (int)(RH / 256), gmax);
  // 6. ht -> LN -> +l_skip -> *right
  out_norm_fused<<<ROWS_, 256, 0, stream>>>(og, p, gmax, gn_g, gn_b, lsk, right, pre);
  // 7. out = pre @ proj_w + proj_b + x_n  (residual fused in epilogue)
  gemm_bf16_wmma<<<gD, 128, 0, stream>>>(pre, H_, proj_w, proj_b, xn, (float*)d_out, D_, D_, H_, 0, 0);
}
